// _PointnetSAModuleBase_13967233646742
// MI455X (gfx1250) — matrix operand order repaired and checked by reading, not compiled, not run on hardware
//
#include <hip/hip_runtime.h>

#define BATCH   8
#define NPTS    16384
#define CIN     64
#define NPOINT  1024
#define NSAMPLE 32
#define RADIUS2 0.01f

typedef __attribute__((ext_vector_type(16))) _Float16 v16h;
typedef __attribute__((ext_vector_type(8)))  _Float16 v8h;
typedef __attribute__((ext_vector_type(8)))  float    v8f;

// ---------------------------------------------------------------------------
// Kernel 1: Furthest Point Sampling. One 1024-thread block per batch.
// Points as float4 in LDS (256KB — needs CDNA5's 320KB WGP LDS) so the
// 1023-step serial chain reads one ds_load_b128 per candidate point.
// ---------------------------------------------------------------------------
__global__ void k_fps(const float* __restrict__ xyz, int* __restrict__ fps_idx) {
    extern __shared__ char smem_raw[];
    float4* pts  = (float4*)smem_raw;            // NPTS float4
    float*  wval = (float*)(pts + NPTS);         // 32 floats
    int*    widx = (int*)(wval + 32);            // 32 ints
    int*    sIdx = widx + 32;                    // 1 int

    const int b    = blockIdx.x;
    const int t    = threadIdx.x;                // 0..1023
    const int lane = t & 31;
    const int wid  = t >> 5;
    const float* base = xyz + (size_t)b * NPTS * 3;

    for (int i = t; i < NPTS; i += 1024)
        pts[i] = make_float4(base[i * 3 + 0], base[i * 3 + 1], base[i * 3 + 2], 0.0f);
    __syncthreads();

    float mind[16];
#pragma unroll
    for (int j = 0; j < 16; ++j) mind[j] = 1e10f;

    int last = 0;
    if (t == 0) fps_idx[b * NPOINT] = 0;

    for (int it = 1; it < NPOINT; ++it) {
        const float4 P = pts[last];              // LDS broadcast
        float bv = -1.0f; int bi = 0;
#pragma unroll
        for (int j = 0; j < 16; ++j) {
            const int i = t + j * 1024;
            const float4 q = pts[i];
            const float dx = q.x - P.x, dy = q.y - P.y, dz = q.z - P.z;
            const float d = dx * dx + dy * dy + dz * dz;
            const float m = fminf(mind[j], d);
            mind[j] = m;
            if (m > bv) { bv = m; bi = i; }
        }
        // wave32 argmax reduction (tie -> lower index, matching argmax)
#pragma unroll
        for (int off = 16; off >= 1; off >>= 1) {
            const float ov = __shfl_xor(bv, off, 32);
            const int   oi = __shfl_xor(bi, off, 32);
            if (ov > bv || (ov == bv && oi < bi)) { bv = ov; bi = oi; }
        }
        if (lane == 0) { wval[wid] = bv; widx[wid] = bi; }
        __syncthreads();
        if (wid == 0) {
            float v  = wval[lane];               // exactly 32 waves
            int   i2 = widx[lane];
#pragma unroll
            for (int off = 16; off >= 1; off >>= 1) {
                const float ov = __shfl_xor(v, off, 32);
                const int   oi = __shfl_xor(i2, off, 32);
                if (ov > v || (ov == v && oi < i2)) { v = ov; i2 = oi; }
            }
            if (lane == 0) { sIdx[0] = i2; fps_idx[b * NPOINT + it] = i2; }
        }
        __syncthreads();
        last = sIdx[0];
    }
}

// ---------------------------------------------------------------------------
// Kernel 2: gather new_xyz / new_normal from fps indices
// ---------------------------------------------------------------------------
__global__ void k_gather(const float* __restrict__ xyz, const float* __restrict__ normal,
                         const int* __restrict__ fps_idx,
                         float* __restrict__ out_xyz, float* __restrict__ out_normal) {
    const int tid = blockIdx.x * blockDim.x + threadIdx.x;
    if (tid >= BATCH * NPOINT) return;
    const int b = tid / NPOINT;
    const int i = fps_idx[tid];
#pragma unroll
    for (int k = 0; k < 3; ++k) {
        out_xyz[tid * 3 + k]    = xyz[((size_t)b * NPTS + i) * 3 + k];
        out_normal[tid * 3 + k] = normal[((size_t)b * NPTS + i) * 3 + k];
    }
}

// ---------------------------------------------------------------------------
// Kernel 3: ball query. One wave32 per center; ballot + prefix popcount to
// append in-ball indices in index order; pad with first in-ball index.
// ---------------------------------------------------------------------------
__global__ void k_ball(const float* __restrict__ xyz, const float* __restrict__ new_xyz,
                       int* __restrict__ ball_idx) {
    const int gwave = (int)((blockIdx.x * blockDim.x + threadIdx.x) >> 5);
    const int lane  = threadIdx.x & 31;
    if (gwave >= BATCH * NPOINT) return;
    const int b = gwave / NPOINT;
    const float cx = new_xyz[gwave * 3 + 0];
    const float cy = new_xyz[gwave * 3 + 1];
    const float cz = new_xyz[gwave * 3 + 2];
    const float* base = xyz + (size_t)b * NPTS * 3;
    int* out = ball_idx + (size_t)gwave * NSAMPLE;

    int cnt = 0;
    int first = 0;
    bool haveFirst = false;
    const unsigned ltmask = (1u << lane) - 1u;

    for (int bi = 0; bi < NPTS; bi += 32) {
        const int i = bi + lane;
        const float dx = base[i * 3 + 0] - cx;
        const float dy = base[i * 3 + 1] - cy;
        const float dz = base[i * 3 + 2] - cz;
        const bool pred = (dx * dx + dy * dy + dz * dz) < RADIUS2;
        const unsigned mask = (unsigned)__ballot(pred);
        if (!haveFirst && mask) { first = bi + (__ffs(mask) - 1); haveFirst = true; }
        if (pred) {
            const int slot = cnt + __popc(mask & ltmask);
            if (slot < NSAMPLE) out[slot] = i;
        }
        cnt += __popc(mask);
        if (cnt >= NSAMPLE) break;   // cnt uniform -> uniform break
    }
    if (cnt < NSAMPLE && lane >= cnt && lane < NSAMPLE) out[lane] = first;
}

// ---------------------------------------------------------------------------
// WMMA fragment helpers — CDNA5 16x16x32 f16 layouts (ISA 7.12.2).
// A (16x32, MxK): lane l holds row M=l&15; elem j -> K = (j>>3)*16 + (l>>4)*8 + (j&7)
//   => two contiguous 8-half (16B) runs per lane: ds_load_b128 x2.
// B fragments are pre-swizzled in LDS so each lane reads 32 contiguous bytes.
// C (16x16 f32) : lane l holds N=l&15; VGPR r -> M = r + (l>>4)*8
// ---------------------------------------------------------------------------
__device__ __forceinline__ v16h pack16(const v8h lo, const v8h hi) {
    return __builtin_shufflevector(lo, hi, 0, 1, 2, 3, 4, 5, 6, 7,
                                           8, 9, 10, 11, 12, 13, 14, 15);
}

__device__ __forceinline__ v16h loadA(const _Float16* X, int ldk, int mt, int kt, int lane) {
    const int m = (lane & 15) + mt * 16;
    const int h = lane >> 4;
    const _Float16* row = X + m * ldk + kt * 32 + h * 8;
    const v8h lo = *(const v8h*)(row);        // K = kt*32 + h*8 + [0..7]
    const v8h hi = *(const v8h*)(row + 16);   // K = kt*32 + 16 + h*8 + [0..7]
    return pack16(lo, hi);
}

__device__ __forceinline__ v16h loadBfrag(const _Float16* Wf, int NT, int nt, int kt, int lane) {
    const _Float16* p = Wf + ((size_t)((kt * NT + nt) * 32 + lane)) * 16;
    const v8h lo = *(const v8h*)(p);
    const v8h hi = *(const v8h*)(p + 8);
    return pack16(lo, hi);
}

__device__ __forceinline__ v8f wmma_acc(const v16h a, const v16h b, const v8f c) {
    return __builtin_amdgcn_wmma_f32_16x16x32_f16(false, a, false, b, (short)0, c,
                                                  false, false);
}

__device__ __forceinline__ void storeC_relu(const v8f c, _Float16* H, int ldn,
                                            int mt, int nt, const float* bias, int lane) {
    const int nl = lane & 15;
    const int h  = lane >> 4;
    const float bb = bias[nt * 16 + nl];
#pragma unroll
    for (int r = 0; r < 8; ++r) {
        const int m = mt * 16 + h * 8 + r;
        H[m * ldn + nt * 16 + nl] = (_Float16)fmaxf(c[r] + bb, 0.0f);
    }
}

// Stage one weight matrix (f32, Kreal x ldn row-major) into LDS as f16
// B-fragments: Wf[((kt*NT + nt)*32 + lane)*16 + j] with K = kt*32 + (j>>3)*16 + (lane>>4)*8 + (j&7).
__device__ __forceinline__ void stageW(const float* __restrict__ W, _Float16* Wf,
                                       int Kreal, int ldn, int KT, int NT, int t) {
    const int total = KT * NT * 512;
    for (int i = t; i < total; i += 256) {
        const int j    = i & 15;
        const int lane = (i >> 4) & 31;
        const int tile = i >> 9;             // kt*NT + nt
        const int nt   = tile % NT;
        const int kt   = tile / NT;
        const int K    = kt * 32 + ((j >> 3) << 4) + ((lane >> 4) << 3) + (j & 7);
        const int n    = nt * 16 + (lane & 15);
        Wf[i] = (K < Kreal) ? (_Float16)W[K * ldn + n] : (_Float16)0.0f;
    }
}

// ---------------------------------------------------------------------------
// Kernel 4: group + 3-layer MLP (WMMA f16->f32) + max-pool over samples.
// 256 threads = 8 waves/block, one center per wave. Dynamic LDS ~149KB.
// ---------------------------------------------------------------------------
__global__ void k_group_mlp(const float* __restrict__ xyz, const float* __restrict__ features,
                            const float* __restrict__ new_xyz, const int* __restrict__ ball_idx,
                            const float* __restrict__ W1, const float* __restrict__ b1,
                            const float* __restrict__ W2, const float* __restrict__ b2,
                            const float* __restrict__ W3, const float* __restrict__ b3,
                            float* __restrict__ out_feat) {
    extern __shared__ char smem_raw[];
    float*    b1s = (float*)smem_raw;          // 64
    float*    b2s = b1s + 64;                  // 64
    float*    b3s = b2s + 64;                  // 128
    _Float16* W1f = (_Float16*)(b3s + 128);    // 3*4*512  (96x64, K padded 67->96)
    _Float16* W2f = W1f + 3 * 4 * 512;         // 2*4*512  (64x64)
    _Float16* W3f = W2f + 2 * 4 * 512;         // 2*8*512  (64x128)
    _Float16* Xt  = W3f + 2 * 8 * 512;         // 8 waves x 32 x 96
    _Float16* H1  = Xt + 8 * 32 * 96;          // 8 waves x 32 x 64
    _Float16* H2  = H1 + 8 * 32 * 64;          // 8 waves x 32 x 64

    const int t = threadIdx.x;

    stageW(W1, W1f, 67, 64, 3, 4, t);
    stageW(W2, W2f, 64, 64, 2, 4, t);
    stageW(W3, W3f, 64, 128, 2, 8, t);
    if (t < 64)  b1s[t] = b1[t];
    if (t < 64)  b2s[t] = b2[t];
    if (t < 128) b3s[t] = b3[t];
    __syncthreads();

    const int lane   = t & 31;
    const int w      = t >> 5;
    const int center = blockIdx.x * 8 + w;      // grid sized exactly
    const int b      = center / NPOINT;

    _Float16* Xw  = Xt + w * 32 * 96;
    _Float16* H1w = H1 + w * 32 * 64;
    _Float16* H2w = H2 + w * 32 * 64;

    // ---- stage grouped input: row = sample (lane), cols = [rel_xyz(3) | feat(64) | pad] ----
    {
        const int p = ball_idx[(size_t)center * NSAMPLE + lane];
        const float* pp = xyz + ((size_t)b * NPTS + p) * 3;
        const float* cc = new_xyz + (size_t)center * 3;
        const float4* ff4 = (const float4*)(features + ((size_t)b * NPTS + p) * CIN);

        _Float16 tmp[96];
        tmp[0] = (_Float16)(pp[0] - cc[0]);
        tmp[1] = (_Float16)(pp[1] - cc[1]);
        tmp[2] = (_Float16)(pp[2] - cc[2]);
#pragma unroll
        for (int c4 = 0; c4 < 16; ++c4) {
            const float4 v = ff4[c4];
            tmp[3 + c4 * 4 + 0] = (_Float16)v.x;
            tmp[3 + c4 * 4 + 1] = (_Float16)v.y;
            tmp[3 + c4 * 4 + 2] = (_Float16)v.z;
            tmp[3 + c4 * 4 + 3] = (_Float16)v.w;
        }
#pragma unroll
        for (int c = 67; c < 96; ++c) tmp[c] = (_Float16)0.0f;

        _Float16* row = Xw + lane * 96;
#pragma unroll
        for (int c = 0; c < 12; ++c)            // 12 x ds_store_b128
            *(v8h*)(row + c * 8) = *(const v8h*)(tmp + c * 8);
    }
    __syncthreads();

    // ---- Layer 1: (32x96) @ (96x64), A fragments hoisted over nt ----
#pragma unroll
    for (int mt = 0; mt < 2; ++mt) {
        const v16h a0 = loadA(Xw, 96, mt, 0, lane);
        const v16h a1 = loadA(Xw, 96, mt, 1, lane);
        const v16h a2 = loadA(Xw, 96, mt, 2, lane);
#pragma unroll
        for (int nt = 0; nt < 4; ++nt) {
            v8f acc = {};
            acc = wmma_acc(a0, loadBfrag(W1f, 4, nt, 0, lane), acc);
            acc = wmma_acc(a1, loadBfrag(W1f, 4, nt, 1, lane), acc);
            acc = wmma_acc(a2, loadBfrag(W1f, 4, nt, 2, lane), acc);
            storeC_relu(acc, H1w, 64, mt, nt, b1s, lane);
        }
    }
    __syncthreads();

    // ---- Layer 2: (32x64) @ (64x64) ----
#pragma unroll
    for (int mt = 0; mt < 2; ++mt) {
        const v16h a0 = loadA(H1w, 64, mt, 0, lane);
        const v16h a1 = loadA(H1w, 64, mt, 1, lane);
#pragma unroll
        for (int nt = 0; nt < 4; ++nt) {
            v8f acc = {};
            acc = wmma_acc(a0, loadBfrag(W2f, 4, nt, 0, lane), acc);
            acc = wmma_acc(a1, loadBfrag(W2f, 4, nt, 1, lane), acc);
            storeC_relu(acc, H2w, 64, mt, nt, b2s, lane);
        }
    }
    __syncthreads();

    // ---- Layer 3: (32x64) @ (64x128), bias+relu+max over 32 samples ----
    const v16h a00 = loadA(H2w, 64, 0, 0, lane);
    const v16h a01 = loadA(H2w, 64, 0, 1, lane);
    const v16h a10 = loadA(H2w, 64, 1, 0, lane);
    const v16h a11 = loadA(H2w, 64, 1, 1, lane);
    float* outp = out_feat + (size_t)center * 128;
#pragma unroll
    for (int nt = 0; nt < 8; ++nt) {
        const v16h bf0 = loadBfrag(W3f, 8, nt, 0, lane);
        const v16h bf1 = loadBfrag(W3f, 8, nt, 1, lane);
        v8f acc0 = {}, acc1 = {};
        acc0 = wmma_acc(a00, bf0, acc0);
        acc0 = wmma_acc(a01, bf1, acc0);
        acc1 = wmma_acc(a10, bf0, acc1);
        acc1 = wmma_acc(a11, bf1, acc1);

        const float bb = b3s[nt * 16 + (lane & 15)];
        float m = 0.0f;  // relu floor
#pragma unroll
        for (int r = 0; r < 8; ++r) {
            m = fmaxf(m, acc0[r] + bb);
            m = fmaxf(m, acc1[r] + bb);
        }
        // lane l covers M halves {h*8..h*8+7, 16+h*8..16+h*8+7}; partner l^16 covers rest
        m = fmaxf(m, __shfl_xor(m, 16, 32));
        if (lane < 16) outp[nt * 16 + lane] = m;
    }
}

// ---------------------------------------------------------------------------
extern "C" void kernel_launch(void* const* d_in, const int* in_sizes, int n_in,
                              void* d_out, int out_size, void* d_ws, size_t ws_size,
                              hipStream_t stream) {
    const float* xyz      = (const float*)d_in[0];
    const float* normal   = (const float*)d_in[1];
    const float* features = (const float*)d_in[2];
    const float* W1 = (const float*)d_in[3];
    const float* b1 = (const float*)d_in[4];
    const float* W2 = (const float*)d_in[5];
    const float* b2 = (const float*)d_in[6];
    const float* W3 = (const float*)d_in[7];
    const float* b3 = (const float*)d_in[8];

    float* out_xyz    = (float*)d_out;
    float* out_normal = out_xyz + BATCH * NPOINT * 3;
    float* out_feat   = out_normal + BATCH * NPOINT * 3;

    int* fps_idx  = (int*)d_ws;                      // 8192 ints
    int* ball_idx = fps_idx + BATCH * NPOINT;        // 262144 ints

    const size_t fps_lds = (size_t)NPTS * sizeof(float4) + 32 * sizeof(float)
                         + 32 * sizeof(int) + 16;                       // ~256.3 KB
    const size_t mlp_lds = 256 * sizeof(float)
                         + (size_t)(3 * 4 * 512 + 2 * 4 * 512 + 2 * 8 * 512) * 2
                         + (size_t)(8 * 32 * 96 + 2 * 8 * 32 * 64) * 2; // ~149 KB

    // CDNA5 WGP has 320KB LDS; raise the dynamic-shared caps.
    (void)hipFuncSetAttribute((const void*)k_fps,
                              hipFuncAttributeMaxDynamicSharedMemorySize, (int)fps_lds);
    (void)hipFuncSetAttribute((const void*)k_group_mlp,
                              hipFuncAttributeMaxDynamicSharedMemorySize, (int)mlp_lds);

    k_fps<<<BATCH, 1024, fps_lds, stream>>>(xyz, fps_idx);
    k_gather<<<(BATCH * NPOINT + 255) / 256, 256, 0, stream>>>(xyz, normal, fps_idx,
                                                               out_xyz, out_normal);
    k_ball<<<(BATCH * NPOINT) / 8, 256, 0, stream>>>(xyz, out_xyz, ball_idx);
    k_group_mlp<<<(BATCH * NPOINT) / 8, 256, mlp_lds, stream>>>(
        xyz, features, out_xyz, ball_idx, W1, b1, W2, b2, W3, b3, out_feat);
}
